// SeqAttnMatch_16209206575368
// MI455X (gfx1250) — matrix operand order repaired and checked by reading, not compiled, not run on hardware
//
#include <hip/hip_runtime.h>

// ---------------------------------------------------------------------------
// SeqAttnMatch for MI455X (gfx1250, wave32, WMMA + async-LDS).
//
// Pipeline (intermediates bf16 in d_ws, ~219 MB, streams through 192MB L2):
//   1) Wbf = bf16(W)                 [H,H]
//   2) yT  = bf16(y)^T               [B,H,L2]  (LDS-tiled transpose)
//   3) xp  = bf16(relu(x W^T + b))   [B,L1,H]  (WMMA bf16)
//   4) yp  = bf16(relu(y W^T + b))   [B,L2,H]
//   5) sc  = mask/bias(xp yp^T)      [B,L1,L2] (WMMA bf16)
//   6) sc  = softmax(sc) in place              (wave32 shfl reduce)
//   7) out = sc @ y  (via yT)        [B,L1,H]  (WMMA bf16, f32 out)
//
// GEMM kernels: block = 8 waves = 128 rows x 64 cols. The 64x32 bf16 B tile
// is staged into double-buffered LDS with GLOBAL_LOAD_ASYNC_TO_LDS_B128
// (ASYNCcnt-tracked). Buffers are published with s_wait_asynccnt 0 followed
// by a raw split barrier (s_barrier_signal/-wait -1) — NOT __syncthreads(),
// whose implicit global fence would stall on the pipelined A-fragment loads
// every k-step. A fragments are register-pipelined one k-step ahead
// (ping-pong, unroll 2, last k-step peeled so the loop body is branch-free).
// ---------------------------------------------------------------------------

#define DEVINL __device__ __forceinline__

typedef __attribute__((ext_vector_type(16))) __bf16 v16bf;
typedef __attribute__((ext_vector_type(8)))  __bf16 v8bf;
typedef __attribute__((ext_vector_type(8)))  float  v8f;
typedef __attribute__((ext_vector_type(4)))  float  v4f;

constexpr int B_  = 32;
constexpr int L1_ = 2048;
constexpr int L2_ = 512;
constexpr int H_  = 768;

constexpr int BSTRIDE = 40;            // LDS row stride in bf16 elems (80B, 16B-aligned)
constexpr int BUFE    = 64 * BSTRIDE;  // one LDS stage buffer, elements

DEVINL v8f wmma_bf16(v16bf a, v16bf b, v8f c) {
  // (neg_a, A, neg_b, B, c_mod, C, reuse_a, reuse_b)
  return __builtin_amdgcn_wmma_f32_16x16x32_bf16(false, a, false, b, (short)0, c,
                                                 false, false);
}

// Async global -> LDS (CDNA5, ASYNCcnt-tracked). lds_off = byte offset within
// the workgroup LDS allocation (low 32 bits of the generic shared address).
DEVINL void async_ld_b128(unsigned lds_off, const void* gaddr) {
  asm volatile("global_load_async_to_lds_b128 %0, %1, off"
               :: "v"(lds_off), "v"((unsigned long long)gaddr) : "memory");
}
DEVINL void wait_async0() {
  asm volatile("s_wait_asynccnt 0x0" ::: "memory");
}
// Raw workgroup split barrier: no implicit global-memory fence (LDS ordering
// is already guaranteed by s_wait_asynccnt for writes and the s_wait_dscnt
// preceding each consuming WMMA for reads).
DEVINL void wg_barrier() {
  asm volatile("s_barrier_signal -1\n\ts_barrier_wait -1" ::: "memory");
}

// --- A fragment pipeline regs (layouts per CDNA5 ISA 7.12.2) ----------------
// A: 16x32 (MxK) bf16. lanes 0-15: row=lane, K halves {k0..k0+7, k0+16..k0+23};
//                      lanes 16-31: row=lane-16, halves {k0+8.., k0+24..}.
struct ABF { v8bf lo, hi; };

DEVINL ABF issue_a_bf16(const __bf16* __restrict__ base, int lda, int k0) {
  const int lane = threadIdx.x & 31;
  const int row  = lane & 15;
  const int koff = k0 + ((lane & 16) ? 8 : 0);
  const __bf16* p = base + (size_t)row * lda + koff;
  ABF r;
  r.lo = *(const v8bf*)(p);
  r.hi = *(const v8bf*)(p + 16);
  return r;
}

DEVINL v16bf combine_a(const ABF& f) {
  v16bf r;
#pragma unroll
  for (int i = 0; i < 8; ++i) { r[i] = f.lo[i]; r[8 + i] = f.hi[i]; }
  return r;
}

struct AF32 { v4f a0, a1, a2, a3; };

DEVINL AF32 issue_a_f32(const float* __restrict__ base, int lda, int k0) {
  const int lane = threadIdx.x & 31;
  const int row  = lane & 15;
  const int koff = k0 + ((lane & 16) ? 8 : 0);
  const float* p = base + (size_t)row * lda + koff;
  AF32 r;
  r.a0 = *(const v4f*)(p);
  r.a1 = *(const v4f*)(p + 4);
  r.a2 = *(const v4f*)(p + 16);
  r.a3 = *(const v4f*)(p + 20);
  return r;
}

DEVINL v16bf cvt_a_f32(const AF32& f) {
  v16bf r;
#pragma unroll
  for (int i = 0; i < 4; ++i) {
    r[i]      = (__bf16)f.a0[i];
    r[4 + i]  = (__bf16)f.a1[i];
    r[8 + i]  = (__bf16)f.a2[i];
    r[12 + i] = (__bf16)f.a3[i];
  }
  return r;
}

// B fragment from the LDS stage: 32x16 (KxN); column n of B = staged row n.
// lanes 0-15: col=lane, K={k0..k0+7, k0+16..k0+23}; lanes 16-31: col=lane-16, K={k0+8.., k0+24..}.
DEVINL v16bf load_b_lds(const __bf16* lbase, int t) {
  const int lane = threadIdx.x & 31;
  const int col  = lane & 15;
  const int koff = (lane & 16) ? 8 : 0;
  const __bf16* p = lbase + (16 * t + col) * BSTRIDE + koff;
  v8bf lo = *(const v8bf*)(p);
  v8bf hi = *(const v8bf*)(p + 16);
  v16bf r;
#pragma unroll
  for (int i = 0; i < 8; ++i) { r[i] = lo[i]; r[8 + i] = hi[i]; }
  return r;
}

// --- prep kernels -----------------------------------------------------------

__global__ void __launch_bounds__(256)
k_cvt_bf16(const float* __restrict__ src, __bf16* __restrict__ dst, int n) {
  int i = (blockIdx.x * blockDim.x + threadIdx.x) * 4;
  if (i + 3 < n) {
    v4f v = *(const v4f*)(src + i);
#pragma unroll
    for (int j = 0; j < 4; ++j) dst[i + j] = (__bf16)v[j];
  }
}

// yT[b][h][m] = bf16(y[b][m][h]), LDS-tiled for coalescing.
__global__ void __launch_bounds__(256)
k_transpose_cvt(const float* __restrict__ y, __bf16* __restrict__ yT) {
  __shared__ float tile[32][33];
  const int b  = blockIdx.z;
  const int m0 = blockIdx.x * 32;  // along L2
  const int h0 = blockIdx.y * 32;  // along H
  const int tx = threadIdx.x, ty = threadIdx.y;  // 32 x 8
  const float* src = y + ((size_t)b * L2_ + m0) * H_ + h0;
#pragma unroll
  for (int i = 0; i < 32; i += 8)
    tile[ty + i][tx] = src[(size_t)(ty + i) * H_ + tx];
  __syncthreads();
  __bf16* dst = yT + ((size_t)b * H_ + h0) * L2_ + m0;
#pragma unroll
  for (int i = 0; i < 32; i += 8)
    dst[(size_t)(ty + i) * L2_ + tx] = (__bf16)tile[tx][ty + i];
}

// --- GEMM kernels -----------------------------------------------------------

// out[r,o] = bf16(relu(sum_h A[r,h]*W[o,h] + bias[o]))   (A fp32)
__global__ void __launch_bounds__(256)
k_proj_relu(const float* __restrict__ A, const __bf16* __restrict__ Wbf,
            const float* __restrict__ bias, __bf16* __restrict__ out) {
  __shared__ __attribute__((aligned(16))) __bf16 lds[2 * BUFE];
  const int wave = threadIdx.x >> 5;
  const int lane = threadIdx.x & 31;
  const size_t r0 = (size_t)blockIdx.x * 128 + wave * 16;
  const int n0 = blockIdx.y * 64;
  const float* Ar = A + r0 * H_;
  // cooperative B-tile loader: thread -> (row, 16B segment)
  const int trow = threadIdx.x >> 2;
  const int tseg = (threadIdx.x & 3) * 8;
  const __bf16* gsrc = Wbf + (size_t)(n0 + trow) * H_ + tseg;
  const unsigned lwr = (unsigned)(size_t)(&lds[trow * BSTRIDE + tseg]);

  async_ld_b128(lwr, gsrc);
  AF32 af[2];
  af[0] = issue_a_f32(Ar, H_, 0);
  wait_async0();
  wg_barrier();

  v8f acc0 = {}, acc1 = {}, acc2 = {}, acc3 = {};
  const int NT = H_ / 32;
#pragma unroll 2
  for (int kt = 0; kt < NT - 1; ++kt) {
    const int nb = (kt + 1) & 1;
    async_ld_b128(lwr + nb * (BUFE * 2), gsrc + (kt + 1) * 32);
    af[nb] = issue_a_f32(Ar, H_, (kt + 1) * 32);
    const __bf16* lcur = &lds[(kt & 1) * BUFE];
    v16bf a = cvt_a_f32(af[kt & 1]);
    acc0 = wmma_bf16(a, load_b_lds(lcur, 0), acc0);
    acc1 = wmma_bf16(a, load_b_lds(lcur, 1), acc1);
    acc2 = wmma_bf16(a, load_b_lds(lcur, 2), acc2);
    acc3 = wmma_bf16(a, load_b_lds(lcur, 3), acc3);
    wait_async0();
    wg_barrier();
  }
  {
    const __bf16* lcur = &lds[((NT - 1) & 1) * BUFE];
    v16bf a = cvt_a_f32(af[(NT - 1) & 1]);
    acc0 = wmma_bf16(a, load_b_lds(lcur, 0), acc0);
    acc1 = wmma_bf16(a, load_b_lds(lcur, 1), acc1);
    acc2 = wmma_bf16(a, load_b_lds(lcur, 2), acc2);
    acc3 = wmma_bf16(a, load_b_lds(lcur, 3), acc3);
  }
  const int col = lane & 15;
  const int rb  = (lane >> 4) * 8;
  v8f accs[4] = {acc0, acc1, acc2, acc3};
#pragma unroll
  for (int t = 0; t < 4; ++t) {
    const int n = n0 + 16 * t + col;
    const float bv = bias[n];
    __bf16* o = out + (r0 + rb) * H_ + n;
#pragma unroll
    for (int j = 0; j < 8; ++j) {
      float v = accs[t][j] + bv;
      o[(size_t)j * H_] = (__bf16)(v > 0.f ? v : 0.f);
    }
  }
}

// sc[b,l,m] = mask ? -inf : (xp[b,l,:].yp[b,m,:] + rw[b,m]*rws)
__global__ void __launch_bounds__(256)
k_scores(const __bf16* __restrict__ xp, const __bf16* __restrict__ yp,
         const int* __restrict__ y_mask, const float* __restrict__ rw,
         const float* __restrict__ rws, __bf16* __restrict__ sc) {
  __shared__ __attribute__((aligned(16))) __bf16 lds[2 * BUFE];
  const int b = blockIdx.z;
  const int wave = threadIdx.x >> 5;
  const int lane = threadIdx.x & 31;
  const size_t l0 = (size_t)blockIdx.x * 128 + wave * 16;  // over L1
  const int m0 = blockIdx.y * 64;                          // over L2
  const __bf16* Ar = xp + ((size_t)b * L1_ + l0) * H_;
  const int trow = threadIdx.x >> 2;
  const int tseg = (threadIdx.x & 3) * 8;
  const __bf16* gsrc = yp + ((size_t)b * L2_ + m0 + trow) * H_ + tseg;
  const unsigned lwr = (unsigned)(size_t)(&lds[trow * BSTRIDE + tseg]);

  async_ld_b128(lwr, gsrc);
  ABF af[2];
  af[0] = issue_a_bf16(Ar, H_, 0);
  wait_async0();
  wg_barrier();

  v8f acc0 = {}, acc1 = {}, acc2 = {}, acc3 = {};
  const int NT = H_ / 32;
#pragma unroll 2
  for (int kt = 0; kt < NT - 1; ++kt) {
    const int nb = (kt + 1) & 1;
    async_ld_b128(lwr + nb * (BUFE * 2), gsrc + (kt + 1) * 32);
    af[nb] = issue_a_bf16(Ar, H_, (kt + 1) * 32);
    const __bf16* lcur = &lds[(kt & 1) * BUFE];
    v16bf a = combine_a(af[kt & 1]);
    acc0 = wmma_bf16(a, load_b_lds(lcur, 0), acc0);
    acc1 = wmma_bf16(a, load_b_lds(lcur, 1), acc1);
    acc2 = wmma_bf16(a, load_b_lds(lcur, 2), acc2);
    acc3 = wmma_bf16(a, load_b_lds(lcur, 3), acc3);
    wait_async0();
    wg_barrier();
  }
  {
    const __bf16* lcur = &lds[((NT - 1) & 1) * BUFE];
    v16bf a = combine_a(af[(NT - 1) & 1]);
    acc0 = wmma_bf16(a, load_b_lds(lcur, 0), acc0);
    acc1 = wmma_bf16(a, load_b_lds(lcur, 1), acc1);
    acc2 = wmma_bf16(a, load_b_lds(lcur, 2), acc2);
    acc3 = wmma_bf16(a, load_b_lds(lcur, 3), acc3);
  }
  const float s = rws[0];
  const int col = lane & 15;
  const int rb  = (lane >> 4) * 8;
  v8f accs[4] = {acc0, acc1, acc2, acc3};
#pragma unroll
  for (int t = 0; t < 4; ++t) {
    const int m = m0 + 16 * t + col;
    const bool msk = y_mask[(size_t)b * L2_ + m] != 0;
    const float add = rw[(size_t)b * L2_ + m] * s;
    __bf16* o = sc + ((size_t)b * L1_ + l0 + rb) * L2_ + m;
#pragma unroll
    for (int j = 0; j < 8; ++j) {
      float v = msk ? -__builtin_inff() : accs[t][j] + add;
      o[(size_t)j * L2_] = (__bf16)v;
    }
  }
}

// in-place softmax over L2 (512) — one wave per row, 16 elems/lane.
__global__ void __launch_bounds__(256)
k_softmax(__bf16* __restrict__ sc) {
  const int wave = threadIdx.x >> 5;
  const int lane = threadIdx.x & 31;
  const size_t row = (size_t)blockIdx.x * 8 + wave;  // over B*L1
  __bf16* p = sc + row * L2_ + lane * 16;
  v8bf c0 = *(const v8bf*)(p);
  v8bf c1 = *(const v8bf*)(p + 8);
  float v[16];
#pragma unroll
  for (int i = 0; i < 8; ++i) { v[i] = (float)c0[i]; v[8 + i] = (float)c1[i]; }
  float m = v[0];
#pragma unroll
  for (int i = 1; i < 16; ++i) m = fmaxf(m, v[i]);
#pragma unroll
  for (int off = 16; off > 0; off >>= 1) m = fmaxf(m, __shfl_xor(m, off, 32));
  float ssum = 0.f;
#pragma unroll
  for (int i = 0; i < 16; ++i) { v[i] = __expf(v[i] - m); ssum += v[i]; }
#pragma unroll
  for (int off = 16; off > 0; off >>= 1) ssum += __shfl_xor(ssum, off, 32);
  const float inv = 1.f / ssum;
#pragma unroll
  for (int i = 0; i < 8; ++i) {
    c0[i] = (__bf16)(v[i] * inv);
    c1[i] = (__bf16)(v[8 + i] * inv);
  }
  *(v8bf*)(p)     = c0;
  *(v8bf*)(p + 8) = c1;
}

// out[b,l,h] = sum_m alpha[b,l,m] * y[b,m,h]   (B operand via yT, f32 store)
__global__ void __launch_bounds__(256)
k_matched(const __bf16* __restrict__ alpha, const __bf16* __restrict__ yT,
          float* __restrict__ out) {
  __shared__ __attribute__((aligned(16))) __bf16 lds[2 * BUFE];
  const int b = blockIdx.z;
  const int wave = threadIdx.x >> 5;
  const int lane = threadIdx.x & 31;
  const size_t l0 = (size_t)blockIdx.x * 128 + wave * 16;  // over L1
  const int h0 = blockIdx.y * 64;                          // over H
  const __bf16* Ar = alpha + ((size_t)b * L1_ + l0) * L2_;
  const int trow = threadIdx.x >> 2;
  const int tseg = (threadIdx.x & 3) * 8;
  const __bf16* gsrc = yT + ((size_t)b * H_ + h0 + trow) * L2_ + tseg;
  const unsigned lwr = (unsigned)(size_t)(&lds[trow * BSTRIDE + tseg]);

  async_ld_b128(lwr, gsrc);
  ABF af[2];
  af[0] = issue_a_bf16(Ar, L2_, 0);
  wait_async0();
  wg_barrier();

  v8f acc0 = {}, acc1 = {}, acc2 = {}, acc3 = {};
  const int NT = L2_ / 32;
#pragma unroll 2
  for (int kt = 0; kt < NT - 1; ++kt) {
    const int nb = (kt + 1) & 1;
    async_ld_b128(lwr + nb * (BUFE * 2), gsrc + (kt + 1) * 32);
    af[nb] = issue_a_bf16(Ar, L2_, (kt + 1) * 32);
    const __bf16* lcur = &lds[(kt & 1) * BUFE];
    v16bf a = combine_a(af[kt & 1]);
    acc0 = wmma_bf16(a, load_b_lds(lcur, 0), acc0);
    acc1 = wmma_bf16(a, load_b_lds(lcur, 1), acc1);
    acc2 = wmma_bf16(a, load_b_lds(lcur, 2), acc2);
    acc3 = wmma_bf16(a, load_b_lds(lcur, 3), acc3);
    wait_async0();
    wg_barrier();
  }
  {
    const __bf16* lcur = &lds[((NT - 1) & 1) * BUFE];
    v16bf a = combine_a(af[(NT - 1) & 1]);
    acc0 = wmma_bf16(a, load_b_lds(lcur, 0), acc0);
    acc1 = wmma_bf16(a, load_b_lds(lcur, 1), acc1);
    acc2 = wmma_bf16(a, load_b_lds(lcur, 2), acc2);
    acc3 = wmma_bf16(a, load_b_lds(lcur, 3), acc3);
  }
  const int col = lane & 15;
  const int rb  = (lane >> 4) * 8;
  v8f accs[4] = {acc0, acc1, acc2, acc3};
#pragma unroll
  for (int t = 0; t < 4; ++t) {
    float* o = out + ((size_t)b * L1_ + l0 + rb) * H_ + h0 + 16 * t + col;
#pragma unroll
    for (int j = 0; j < 8; ++j) o[(size_t)j * H_] = accs[t][j];
  }
}

// --- host launcher ----------------------------------------------------------

extern "C" void kernel_launch(void* const* d_in, const int* in_sizes, int n_in,
                              void* d_out, int out_size, void* d_ws, size_t ws_size,
                              hipStream_t stream) {
  const float* x        = (const float*)d_in[0];
  const float* y        = (const float*)d_in[1];
  const int*   y_mask   = (const int*)d_in[2];
  const float* rweights = (const float*)d_in[3];
  const float* W        = (const float*)d_in[4];
  const float* bias     = (const float*)d_in[5];
  const float* rscalar  = (const float*)d_in[6];
  float* out = (float*)d_out;

  // workspace layout (all sizes multiples of 256B)
  char* ws = (char*)d_ws;
  const size_t WBF_BYTES = (size_t)H_ * H_ * 2;          //   1.1 MB
  const size_t XP_BYTES  = (size_t)B_ * L1_ * H_ * 2;    // 100.7 MB
  const size_t YP_BYTES  = (size_t)B_ * L2_ * H_ * 2;    //  25.2 MB
  const size_t YT_BYTES  = (size_t)B_ * H_ * L2_ * 2;    //  25.2 MB
  __bf16* Wbf = (__bf16*)(ws);
  __bf16* xp  = (__bf16*)(ws + WBF_BYTES);
  __bf16* yp  = (__bf16*)(ws + WBF_BYTES + XP_BYTES);
  __bf16* yT  = (__bf16*)(ws + WBF_BYTES + XP_BYTES + YP_BYTES);
  __bf16* sc  = (__bf16*)(ws + WBF_BYTES + XP_BYTES + YP_BYTES + YT_BYTES);

  // 1) W -> bf16
  k_cvt_bf16<<<dim3((H_ * H_) / 1024), dim3(256), 0, stream>>>(W, Wbf, H_ * H_);
  // 2) y -> yT bf16
  k_transpose_cvt<<<dim3(L2_ / 32, H_ / 32, B_), dim3(32, 8), 0, stream>>>(y, yT);
  // 3) x_proj
  k_proj_relu<<<dim3((B_ * L1_) / 128, H_ / 64), dim3(256), 0, stream>>>(x, Wbf, bias, xp);
  // 4) y_proj
  k_proj_relu<<<dim3((B_ * L2_) / 128, H_ / 64), dim3(256), 0, stream>>>(y, Wbf, bias, yp);
  // 5) scores + mask + recency bias
  k_scores<<<dim3(L1_ / 128, L2_ / 64, B_), dim3(256), 0, stream>>>(
      xp, yp, y_mask, rweights, rscalar, sc);
  // 6) softmax in place
  k_softmax<<<dim3((B_ * L1_) / 8), dim3(256), 0, stream>>>(sc);
  // 7) matched = alpha @ y
  k_matched<<<dim3(L1_ / 128, H_ / 64, B_), dim3(256), 0, stream>>>(sc, yT, out);
}
